// SA_GC_9259949490995
// MI455X (gfx1250) — matrix operand order repaired and checked by reading, not compiled, not run on hardware
//
#include <hip/hip_runtime.h>
#include <hip/hip_bf16.h>

// SA-GC fused kernel for MI455X (gfx1250), wave32 + WMMA f16->f32.
// One block handles 4 consecutive t for one n: loads x slice once (residual +
// feat), runs LN -> qk GEMM -> dots -> softmax*topo -> z GEMM -> conv GEMM ->
// BN+res+relu, all matmuls via v_wmma_f32_16x16x32_f16 on LDS-staged f16 tiles.

typedef _Float16 f16_t;
typedef __attribute__((ext_vector_type(16))) _Float16 v16h;
typedef __attribute__((ext_vector_type(8)))  _Float16 v8h;
typedef __attribute__((ext_vector_type(8)))  float    v8f;

#define N_    64
#define C_    128
#define T_    256
#define V_    25
#define H_    2
#define TV_   (T_*V_)
#define INNER 84      // rel_channels * heads
#define QKOUT 168     // 2*INNER
#define QKPAD 176     // padded to 11 col tiles of 16
#define DPAD  96      // inner padded to 3 K-steps of 32
#define VP    32      // V padded
#define BT    4       // t-values per block (coalescing)
#define NTH   256

// workspace byte offsets
#define WS_WQKH   0          // 176*128 f16 (rows>=168 zero)
#define WS_CONVWH 45056      // 2*128*128 f16
#define WS_BIAS   110592     // 128 f32 : sum_h conv_b
#define WS_SCALE  111104     // 128 f32 : bn_gamma * rsqrt(1+eps)
#define WS_SHIFT  111616     // 128 f32 : bn_beta
#define SMEM_BYTES 155648

__device__ __forceinline__ v16h cat8(v8h lo, v8h hi) {
  return __builtin_shufflevector(lo, hi, 0,1,2,3,4,5,6,7,8,9,10,11,12,13,14,15);
}
// A operand (16xK row-major in LDS): lane holds row M=lane%16; g=lane/16 picks
// 16 consecutive K [KB+g*16 .. +15] (matches B operand order).
__device__ __forceinline__ v16h ldsA_frag(const f16_t* rowPtr, int KB, int g) {
  v8h lo = *(const v8h*)(rowPtr + KB + g*16);
  v8h hi = *(const v8h*)(rowPtr + KB + g*16 + 8);
  return cat8(lo, hi);
}
// B operand: lane holds column N=lane%16, 16 consecutive K at KB+g*16 (32B).
__device__ __forceinline__ v16h ldB_frag(const f16_t* p) { return *(const v16h*)p; }

#define WMMA(a,b,c) __builtin_amdgcn_wmma_f32_16x16x32_f16(false,(a),false,(b),(short)0,(c),false,false)

__global__ __launch_bounds__(NTH)
void sagc_prep(const float* __restrict__ Wqk, const float* __restrict__ conv_w,
               const float* __restrict__ conv_b, const float* __restrict__ bn_g,
               const float* __restrict__ bn_b,
               f16_t* __restrict__ wqkh, f16_t* __restrict__ convwh,
               float* __restrict__ bias, float* __restrict__ scale,
               float* __restrict__ shift) {
  int i = blockIdx.x * NTH + threadIdx.x;
  if (i < QKPAD * C_) {
    int o = i / C_;
    wqkh[i] = (o < QKOUT) ? (f16_t)Wqk[(size_t)o * C_ + (i % C_)] : (f16_t)0.f;
  }
  if (i < H_ * C_ * C_) convwh[i] = (f16_t)conv_w[i];
  if (i < C_) {
    bias[i]  = conv_b[i] + conv_b[C_ + i];
    scale[i] = bn_g[i] * rsqrtf(1.f + 1e-5f);
    shift[i] = bn_b[i];
  }
}

__global__ __launch_bounds__(NTH)
void sagc_main(const float* __restrict__ x, const float* __restrict__ topo,
               const float* __restrict__ lnG, const float* __restrict__ lnB,
               const float* __restrict__ bqk,
               const f16_t* __restrict__ wqkh, const f16_t* __restrict__ convwh,
               const float* __restrict__ bias, const float* __restrict__ scale,
               const float* __restrict__ shift, float* __restrict__ out) {
  extern __shared__ char smem[];
  float* sX    = (float*)smem;            // [128][100] raw y slice (fp32)
  float* sOut  = sX + C_ * BT * V_;       // [128][100] result buffer
  float* sDots = sOut + C_ * BT * V_;     // [32][32]   attention logits/probs
  f16_t* sYn   = (f16_t*)(sDots + VP*VP); // [32][128]  layernormed (f16)
  f16_t* sYT   = sYn + VP * C_;           // [128][32]  y transposed (f16)
  f16_t* sQ    = sYT + C_ * VP;           // [32][96]
  f16_t* sK    = sQ + VP * DPAD;          // [32][96]
  f16_t* sA    = sK + VP * DPAD;          // [2][32][32] attn*topo (f16)
  f16_t* sZ    = sA + H_ * VP * VP;       // [2][32][128]

  const int tid  = threadIdx.x;
  const int lane = tid & 31, wave = tid >> 5;
  const int g = lane >> 4, ln = lane & 15;
  const int n  = blockIdx.x >> 6;          // 64 t-groups per n
  const int t0 = (blockIdx.x & 63) * BT;

  // ---- load x[n, :, t0:t0+4, :] : contiguous 400B runs per channel ----
  const float* xb = x + (size_t)n * C_ * TV_ + (size_t)t0 * V_;
  for (int i = tid; i < C_ * BT * V_; i += NTH) {
    int c = i / (BT * V_), r = i % (BT * V_);
    sX[c * (BT * V_) + r] = xb[(size_t)c * TV_ + r];
  }
  __syncthreads();

  for (int bt = 0; bt < BT; ++bt) {
    // ---- zero all f16 staging (sYn..sA contiguous: 16384 halfs) ----
    {
      uint32_t* zp = (uint32_t*)sYn;
      for (int i = tid; i < 8192; i += NTH) zp[i] = 0u;
    }
    __syncthreads();

    // ---- LayerNorm over C per joint row; also build y^T f16 for z GEMM ----
    for (int v = wave; v < V_; v += 8) {
      float vals[4], s = 0.f, ss = 0.f;
      for (int k = 0; k < 4; ++k) {
        int c = lane + 32 * k;
        float f = sX[c * (BT * V_) + bt * V_ + v];
        vals[k] = f; s += f; ss += f * f;
      }
      for (int off = 16; off > 0; off >>= 1) {
        s  += __shfl_xor(s,  off, 32);
        ss += __shfl_xor(ss, off, 32);
      }
      float mu   = s * (1.f / 128.f);
      float var  = ss * (1.f / 128.f) - mu * mu;
      float rstd = rsqrtf(var + 1e-5f);
      for (int k = 0; k < 4; ++k) {
        int c = lane + 32 * k;
        sYn[v * C_ + c] = (f16_t)((vals[k] - mu) * rstd * lnG[c] + lnB[c]);
        sYT[c * VP + v] = (f16_t)vals[k];
      }
    }
    __syncthreads();

    // ---- qk GEMM: (32x128) x (128x176) -> q|k, 22 tiles over 8 waves ----
    for (int idt = wave; idt < 22; idt += 8) {
      int rt = idt / 11, ct = idt % 11;
      v8f acc = {};
      const f16_t* arow = sYn + (rt * 16 + ln) * C_;
      const f16_t* brow = wqkh + (size_t)(ct * 16 + ln) * C_;
      for (int KB = 0; KB < C_; KB += 32)
        acc = WMMA(ldsA_frag(arow, KB, g), ldB_frag(brow + KB + g * 16), acc);
      int o = ct * 16 + ln;
      float bval = (o < QKOUT) ? bqk[o] : 0.f;
      for (int r = 0; r < 8; ++r) {
        int v = rt * 16 + g * 8 + r;
        float val = acc[r] + bval;
        if (o < INNER)      sQ[v * DPAD + o]           = (f16_t)val;
        else if (o < QKOUT) sK[v * DPAD + (o - INNER)] = (f16_t)val;
      }
    }
    __syncthreads();

    // ---- dots = q k^T * hidden^-0.5 : 2x2 tiles, K padded 84->96 ----
    if (wave < 4) {
      int rt = wave >> 1, ct = wave & 1;
      v8f acc = {};
      const f16_t* arow = sQ + (rt * 16 + ln) * DPAD;
      const f16_t* brow = sK + (ct * 16 + ln) * DPAD;
      for (int KB = 0; KB < DPAD; KB += 32)
        acc = WMMA(ldsA_frag(arow, KB, g), ldB_frag(brow + KB + g * 16), acc);
      for (int r = 0; r < 8; ++r) {
        int v = rt * 16 + g * 8 + r;
        sDots[v * VP + ct * 16 + ln] = acc[r] * 0.154303349962092f; // 42^-0.5
      }
    }
    __syncthreads();

    // ---- softmax rows (<25 cols) and per-head topology mask ----
    if (tid < V_) {
      int v = tid;
      float m = -1e30f;
      for (int j = 0; j < V_; ++j) m = fmaxf(m, sDots[v * VP + j]);
      float ssum = 0.f;
      for (int j = 0; j < V_; ++j) {
        float e = __expf(sDots[v * VP + j] - m);
        sDots[v * VP + j] = e; ssum += e;
      }
      float inv = 1.f / ssum;
      for (int j = 0; j < V_; ++j) {
        float a = sDots[v * VP + j] * inv;
        sA[0 * VP * VP + v * VP + j] = (f16_t)(a * topo[0 * V_ * V_ + v * V_ + j]);
        sA[1 * VP * VP + v * VP + j] = (f16_t)(a * topo[1 * V_ * V_ + v * V_ + j]);
      }
    }
    __syncthreads();

    // ---- z = A_h @ feat : 2h x 2rt x 8ct tiles, single K-step (VP=32) ----
    for (int idt = wave; idt < 32; idt += 8) {
      int h = idt >> 4, rt = (idt >> 3) & 1, ct = idt & 7;
      v8f acc = {};
      const f16_t* arow = sA + h * VP * VP + (rt * 16 + ln) * VP;
      const f16_t* brow = sYT + (ct * 16 + ln) * VP;
      acc = WMMA(ldsA_frag(arow, 0, g), ldB_frag(brow + g * 16), acc);
      for (int r = 0; r < 8; ++r) {
        int v = rt * 16 + g * 8 + r;
        sZ[h * VP * C_ + v * C_ + ct * 16 + ln] = (f16_t)acc[r];
      }
    }
    __syncthreads();

    // ---- out = sum_h z_h @ W_h^T, then BN + residual + relu ----
    for (int idt = wave; idt < 16; idt += 8) {
      int rt = idt >> 3, ct = idt & 7;
      v8f acc = {};
      for (int h = 0; h < H_; ++h) {
        const f16_t* arow = sZ + h * VP * C_ + (rt * 16 + ln) * C_;
        const f16_t* brow = convwh + ((size_t)h * C_ + ct * 16 + ln) * C_;
        for (int KB = 0; KB < C_; KB += 32)
          acc = WMMA(ldsA_frag(arow, KB, g), ldB_frag(brow + KB + g * 16), acc);
      }
      int o = ct * 16 + ln;
      float sc = scale[o], sh = shift[o], bi = bias[o];
      for (int r = 0; r < 8; ++r) {
        int v = rt * 16 + g * 8 + r;
        if (v < V_) {
          float val = (acc[r] + bi) * sc + sh + sX[o * (BT * V_) + bt * V_ + v];
          sOut[o * (BT * V_) + bt * V_ + v] = fmaxf(val, 0.f);
        }
      }
    }
    __syncthreads();
  }

  // ---- coalesced store: 400B contiguous per channel ----
  float* ob = out + (size_t)n * C_ * TV_ + (size_t)t0 * V_;
  for (int i = tid; i < C_ * BT * V_; i += NTH) {
    int c = i / (BT * V_), r = i % (BT * V_);
    ob[(size_t)c * TV_ + r] = sOut[i];
  }
}

extern "C" void kernel_launch(void* const* d_in, const int* in_sizes, int n_in,
                              void* d_out, int out_size, void* d_ws, size_t ws_size,
                              hipStream_t stream) {
  const float* x     = (const float*)d_in[0];
  const float* topo  = (const float*)d_in[1];
  const float* lnG   = (const float*)d_in[2];
  const float* lnB   = (const float*)d_in[3];
  const float* Wqk   = (const float*)d_in[4];
  const float* bqk   = (const float*)d_in[5];
  const float* convw = (const float*)d_in[6];
  const float* convb = (const float*)d_in[7];
  const float* bng   = (const float*)d_in[8];
  const float* bnb   = (const float*)d_in[9];
  float* outp = (float*)d_out;

  f16_t* wqkh   = (f16_t*)((char*)d_ws + WS_WQKH);
  f16_t* convwh = (f16_t*)((char*)d_ws + WS_CONVWH);
  float* bias   = (float*)((char*)d_ws + WS_BIAS);
  float* scale  = (float*)((char*)d_ws + WS_SCALE);
  float* shift  = (float*)((char*)d_ws + WS_SHIFT);

  (void)in_sizes; (void)n_in; (void)out_size; (void)ws_size;
  hipFuncSetAttribute((const void*)sagc_main,
                      hipFuncAttributeMaxDynamicSharedMemorySize, SMEM_BYTES);

  sagc_prep<<<(H_ * C_ * C_ + NTH - 1) / NTH, NTH, 0, stream>>>(
      Wqk, convw, convb, bng, bnb, wqkh, convwh, bias, scale, shift);
  sagc_main<<<dim3(N_ * (T_ / BT)), dim3(NTH), SMEM_BYTES, stream>>>(
      x, topo, lnG, lnB, bqk, wqkh, convwh, bias, scale, shift, outp);
}
